// NeuralMessagePasser_30537217474923
// MI455X (gfx1250) — compile-verified, matrix operand order checked by reading, not run on hardware
//
#include <hip/hip_runtime.h>
#include <stdint.h>

// ---------------- problem constants ----------------
#define E_EDGES   8192
#define V_NODES   2048
#define F_NODES   2048
#define B_BATCH   8
#define H_DIM     128
#define DEC_DIM   128
#define EDGE_DIM  2
#define META_DIM  4
#define MEMH      256
#define MEMAGG    128
#define AGGH      128
#define IN_DIM    134      // DEC + EDGE + META
#define K1PAD     160      // 134 padded to multiple of 32
#define K3PAD     160      // 130 padded to multiple of 32

typedef __attribute__((ext_vector_type(16))) __bf16 v16bf;
typedef __attribute__((ext_vector_type(8)))  float  v8f;

static __device__ __forceinline__ unsigned short f2bf(float f) {
  union { float f; unsigned u; } x; x.f = f;
  unsigned r = x.u + 0x7FFFu + ((x.u >> 16) & 1u);   // round-to-nearest-even
  return (unsigned short)(r >> 16);
}

// ---------- index / broadcast recovery from one-hot masks ----------
__global__ void node_batch_kernel(const float* __restrict__ bvm,   // (V,B)
                                  const int*   __restrict__ act,   // (B,1) int
                                  const float* __restrict__ meta,  // (B,4)
                                  float* __restrict__ var_act,     // (V)
                                  float* __restrict__ var_meta) {  // (V,4)
  int v = blockIdx.x * blockDim.x + threadIdx.x;
  if (v >= V_NODES) return;
  int b = 0;
  for (int j = 0; j < B_BATCH; ++j)
    if (bvm[v * B_BATCH + j] != 0.f) b = j;
  var_act[v] = (float)act[b];
  for (int c = 0; c < META_DIM; ++c)
    var_meta[v * META_DIM + c] = meta[b * META_DIM + c];
}

// one wave scans one (E x n) one-hot row; wave32 max-reduction
__global__ void edge_index_kernel(const float* __restrict__ mask_t, int n,
                                  int* __restrict__ idx) {
  int wave = blockIdx.x * (blockDim.x >> 5) + (threadIdx.x >> 5);
  int lane = threadIdx.x & 31;
  if (wave >= E_EDGES) return;
  const float* row = mask_t + (size_t)wave * n;
  int found = 0;
  for (int j = lane; j < n; j += 32)
    if (row[j] != 0.f) found = j;
  for (int off = 16; off > 0; off >>= 1) {
    int o = __shfl_xor(found, off, 32);
    found = found > o ? found : o;
  }
  if (lane == 0) idx[wave] = found;
}

__global__ void edge_gather_kernel(const int* __restrict__ vidx,
                                   const float* __restrict__ var_act,
                                   const float* __restrict__ var_meta,
                                   float* __restrict__ mask_e,   // (E)
                                   float* __restrict__ gfeat) {  // (E,4)
  int e = blockIdx.x * blockDim.x + threadIdx.x;
  if (e >= E_EDGES) return;
  int v = vidx[e];
  mask_e[e] = var_act[v];
  for (int c = 0; c < META_DIM; ++c)
    gfeat[e * META_DIM + c] = var_meta[v * META_DIM + c];
}

// ---------------- bf16 packing kernels ----------------
__global__ void pack_x1_kernel(const float* __restrict__ dec,   // (E,128)
                               const float* __restrict__ ef,    // (E,2)
                               const float* __restrict__ gf,    // (E,4)
                               unsigned short* __restrict__ X)  // (E,K1PAD)
{
  int t = blockIdx.x * blockDim.x + threadIdx.x;
  if (t >= E_EDGES * K1PAD) return;
  int e = t / K1PAD, k = t - e * K1PAD;
  float v = 0.f;
  if (k < DEC_DIM)                     v = dec[e * DEC_DIM + k];
  else if (k < DEC_DIM + EDGE_DIM)     v = ef[e * EDGE_DIM + (k - DEC_DIM)];
  else if (k < IN_DIM)                 v = gf[e * META_DIM + (k - DEC_DIM - EDGE_DIM)];
  X[t] = f2bf(v);
}

// W (K,N) f32 row-major  ->  Wt (N,Kpad) bf16 (K-major per output column)
__global__ void pack_w_kernel(const float* __restrict__ W,
                              int K, int Kpad, int N,
                              unsigned short* __restrict__ Wt) {
  int t = blockIdx.x * blockDim.x + threadIdx.x;
  if (t >= N * Kpad) return;
  int nn = t / Kpad, k = t - nn * Kpad;
  float v = (k < K) ? W[k * N + nn] : 0.f;
  Wt[t] = f2bf(v);
}

__global__ void zero_kernel(float* __restrict__ p, int n) {
  int t = blockIdx.x * blockDim.x + threadIdx.x;
  if (t < n) p[t] = 0.f;
}

__global__ void scatter_add_kernel(const int* __restrict__ idx,
                                   const float* __restrict__ h,  // (E,128)
                                   float* __restrict__ S) {      // (nodes,128)
  int t = blockIdx.x * blockDim.x + threadIdx.x;
  if (t >= E_EDGES * MEMAGG) return;
  int e = t / MEMAGG, c = t - e * MEMAGG;
  atomicAdd(&S[idx[e] * MEMAGG + c], h[t]);
}

__global__ void pack_x3_kernel(const int* __restrict__ idx,
                               const float* __restrict__ S,     // (nodes,128)
                               const float* __restrict__ h,     // (E,128)
                               const float* __restrict__ ef,    // (E,2)
                               unsigned short* __restrict__ X)  // (E,K3PAD)
{
  int t = blockIdx.x * blockDim.x + threadIdx.x;
  if (t >= E_EDGES * K3PAD) return;
  int e = t / K3PAD, k = t - e * K3PAD;
  float v = 0.f;
  if (k < MEMAGG)                  v = S[idx[e] * MEMAGG + k] - h[e * MEMAGG + k];
  else if (k < MEMAGG + EDGE_DIM)  v = ef[e * EDGE_DIM + (k - MEMAGG)];
  X[t] = f2bf(v);
}

__global__ void blend_kernel(const float* __restrict__ agg,    // (E,128)
                             const float* __restrict__ oldst,  // (E,128)
                             const float* __restrict__ mask_e, // (E)
                             float* __restrict__ out) {
  int t = blockIdx.x * blockDim.x + threadIdx.x;
  if (t >= E_EDGES * H_DIM) return;
  int e = t / H_DIM;
  float m = mask_e[e];
  out[t] = m * agg[t] + (1.f - m) * oldst[t];
}

// ---------------- WMMA bf16 GEMM, register-blocked 16x64 per wave ----------------
// X  : (M,Kpad) bf16 row-major (activations)
// Wt : (N,Kpad) bf16 row-major (weights, K-major per column)
// One wave computes a 16(M) x 64(N) strip: the A fragment is loaded once per
// 32-deep K step and fed to 4 v_wmma_f32_16x16x32_bf16 against 4 B fragments.
// N must be a multiple of 64 (256 / 128 here).
__global__ void __launch_bounds__(256)
gemm_bf16_wmma(const unsigned short* __restrict__ X,
               const unsigned short* __restrict__ Wt,
               const float* __restrict__ bias,
               float* __restrict__ outF,            // (M,N) f32 or null
               unsigned short* __restrict__ outBf,  // (M,N) bf16 or null
               int M, int Kpad, int N, int doRelu) {
  int wave = blockIdx.x * (blockDim.x >> 5) + (threadIdx.x >> 5);
  int lane = threadIdx.x & 31;
  int nGroups = N >> 6;                 // groups of 4 n-tiles (64 cols)
  int mTiles  = M >> 4;
  int mTile = wave / nGroups;
  int group = wave - mTile * nGroups;
  if (mTile >= mTiles) return;          // wave-uniform: EXEC all-ones at WMMA

  const int r16 = lane & 15;            // A row / B col within tile
  const int kh  = lane >> 4;            // K-half select (ISA 7.12.2 16-bit layout)
  const unsigned short* arow = X + (size_t)(mTile * 16 + r16) * Kpad;
  const unsigned short* brow0 = Wt + (size_t)(group * 64 + r16) * Kpad;

  v8f acc0 = {}, acc1 = {}, acc2 = {}, acc3 = {};
  for (int k0 = 0; k0 < Kpad; k0 += 32) {
    __builtin_prefetch(arow + k0 + 64, 0, 1);   // global_prefetch_b8
    union { unsigned u[8]; v16bf v; } fa, fb0, fb1, fb2, fb3;
#pragma unroll
    for (int r = 0; r < 8; ++r) {
      int k = k0 + ((r < 4) ? (8 * kh + 2 * r) : (16 + 8 * kh + 2 * (r - 4)));
      fa.u[r]  = *(const unsigned*)(arow + k);
      fb0.u[r] = *(const unsigned*)(brow0 + k);
      fb1.u[r] = *(const unsigned*)(brow0 + 16 * Kpad + k);
      fb2.u[r] = *(const unsigned*)(brow0 + 32 * Kpad + k);
      fb3.u[r] = *(const unsigned*)(brow0 + 48 * Kpad + k);
    }
    acc0 = __builtin_amdgcn_wmma_f32_16x16x32_bf16(false, fa.v, false, fb0.v,
                                                   (short)0, acc0, false, false);
    acc1 = __builtin_amdgcn_wmma_f32_16x16x32_bf16(false, fa.v, false, fb1.v,
                                                   (short)0, acc1, false, false);
    acc2 = __builtin_amdgcn_wmma_f32_16x16x32_bf16(false, fa.v, false, fb2.v,
                                                   (short)0, acc2, false, false);
    acc3 = __builtin_amdgcn_wmma_f32_16x16x32_bf16(false, fa.v, false, fb3.v,
                                                   (short)0, acc3, false, false);
  }

  v8f accs[4] = {acc0, acc1, acc2, acc3};
#pragma unroll
  for (int t = 0; t < 4; ++t) {
    int col = group * 64 + t * 16 + r16;
    float bv = bias ? bias[col] : 0.f;
#pragma unroll
    for (int r = 0; r < 8; ++r) {
      int row = mTile * 16 + r + 8 * kh;  // C/D layout: VGPR r -> M = r + 8*half
      float v = accs[t][r] + bv;
      if (doRelu) v = v > 0.f ? v : 0.f;
      size_t o = (size_t)row * N + col;
      if (outF)  outF[o]  = v;
      if (outBf) outBf[o] = f2bf(v);
    }
  }
}

// ---------------- host orchestration ----------------
extern "C" void kernel_launch(void* const* d_in, const int* in_sizes, int n_in,
                              void* d_out, int out_size, void* d_ws, size_t ws_size,
                              hipStream_t stream) {
  const float* variable_state  = (const float*)d_in[0];
  const float* function_state  = (const float*)d_in[1];
  const float* dec_var         = (const float*)d_in[2];
  const float* dec_fun         = (const float*)d_in[3];
  const float* edge_feature    = (const float*)d_in[4];
  const float* meta_data       = (const float*)d_in[5];
  const int*   active_mask     = (const int*)d_in[6];
  const float* variable_mask_t = (const float*)d_in[8];
  const float* function_mask_t = (const float*)d_in[10];
  const float* b_variable_mask = (const float*)d_in[11];
  // weights: v branch 12..19, f branch 20..27 (M1w,M1b,M2w,M2b,A1w,A1b,A2w,A2b)

  char* ws = (char*)d_ws;
  size_t off = 0;
  auto take = [&](size_t bytes) -> char* {
    char* p = ws + off;
    off = (off + bytes + 255) & ~(size_t)255;
    return p;
  };

  int*   var_idx  = (int*)  take((size_t)E_EDGES * 4);
  int*   fun_idx  = (int*)  take((size_t)E_EDGES * 4);
  float* var_act  = (float*)take((size_t)V_NODES * 4);
  float* var_meta = (float*)take((size_t)V_NODES * META_DIM * 4);
  float* mask_e   = (float*)take((size_t)E_EDGES * 4);
  float* gfeat    = (float*)take((size_t)E_EDGES * META_DIM * 4);
  unsigned short* X1 = (unsigned short*)take((size_t)E_EDGES * K1PAD * 2);
  unsigned short* X2 = (unsigned short*)take((size_t)E_EDGES * MEMH * 2);
  float*          H2 = (float*)         take((size_t)E_EDGES * MEMAGG * 4);
  float*          S  = (float*)         take((size_t)V_NODES * MEMAGG * 4);
  unsigned short* X3 = (unsigned short*)take((size_t)E_EDGES * K3PAD * 2);
  unsigned short* X4 = (unsigned short*)take((size_t)E_EDGES * AGGH * 2);
  float*          AGG= (float*)         take((size_t)E_EDGES * H_DIM * 4);
  unsigned short* W1t= (unsigned short*)take((size_t)MEMH  * K1PAD * 2);
  unsigned short* W2t= (unsigned short*)take((size_t)MEMAGG* MEMH  * 2);
  unsigned short* A1t= (unsigned short*)take((size_t)AGGH  * K3PAD * 2);
  unsigned short* A2t= (unsigned short*)take((size_t)H_DIM * AGGH  * 2);

  auto cdiv = [](int a, int b) { return (a + b - 1) / b; };
  const int BT = 256;

  // ---- shared precompute: indices, per-edge mask and meta broadcast ----
  node_batch_kernel<<<cdiv(V_NODES, BT), BT, 0, stream>>>(
      b_variable_mask, active_mask, meta_data, var_act, var_meta);
  edge_index_kernel<<<cdiv(E_EDGES, 8), BT, 0, stream>>>(variable_mask_t, V_NODES, var_idx);
  edge_index_kernel<<<cdiv(E_EDGES, 8), BT, 0, stream>>>(function_mask_t, F_NODES, fun_idx);
  edge_gather_kernel<<<cdiv(E_EDGES, BT), BT, 0, stream>>>(
      var_idx, var_act, var_meta, mask_e, gfeat);

  float* out = (float*)d_out;

  auto run_branch = [&](const float* dec, const int* idx,
                        const float* W1, const float* b1,
                        const float* W2, const float* b2,
                        const float* A1, const float* c1,
                        const float* A2, const float* c2,
                        const float* oldst, float* outHalf) {
    // weight packing (tiny)
    pack_w_kernel<<<cdiv(MEMH  * K1PAD, BT), BT, 0, stream>>>(W1, IN_DIM,           K1PAD, MEMH,  W1t);
    pack_w_kernel<<<cdiv(MEMAGG* MEMH , BT), BT, 0, stream>>>(W2, MEMH,             MEMH,  MEMAGG,W2t);
    pack_w_kernel<<<cdiv(AGGH  * K3PAD, BT), BT, 0, stream>>>(A1, MEMAGG + EDGE_DIM,K3PAD, AGGH,  A1t);
    pack_w_kernel<<<cdiv(H_DIM * AGGH , BT), BT, 0, stream>>>(A2, AGGH,             AGGH,  H_DIM, A2t);

    // X1 = concat(dec_state, edge_feature, graph_feat) -> bf16, padded
    pack_x1_kernel<<<cdiv(E_EDGES * K1PAD, BT), BT, 0, stream>>>(dec, edge_feature, gfeat, X1);

    // G1: (E,160)x(160,256) + b1, ReLU -> X2 (bf16)
    {
      int waves = (E_EDGES / 16) * (MEMH / 64);
      gemm_bf16_wmma<<<cdiv(waves, 8), BT, 0, stream>>>(
          X1, W1t, b1, (float*)nullptr, X2, E_EDGES, K1PAD, MEMH, 1);
    }
    // G2: (E,256)x(256,128) + b2 -> H2 (f32)
    {
      int waves = (E_EDGES / 16) * (MEMAGG / 64);
      gemm_bf16_wmma<<<cdiv(waves, 8), BT, 0, stream>>>(
          X2, W2t, b2, H2, (unsigned short*)nullptr, E_EDGES, MEMH, MEMAGG, 0);
    }

    // segment-sum per node, then agg[e] = S[idx[e]] - h[e]
    zero_kernel<<<cdiv(V_NODES * MEMAGG, BT), BT, 0, stream>>>(S, V_NODES * MEMAGG);
    scatter_add_kernel<<<cdiv(E_EDGES * MEMAGG, BT), BT, 0, stream>>>(idx, H2, S);
    pack_x3_kernel<<<cdiv(E_EDGES * K3PAD, BT), BT, 0, stream>>>(idx, S, H2, edge_feature, X3);

    // G3: (E,160)x(160,128) + c1, ReLU -> X4 (bf16)
    {
      int waves = (E_EDGES / 16) * (AGGH / 64);
      gemm_bf16_wmma<<<cdiv(waves, 8), BT, 0, stream>>>(
          X3, A1t, c1, (float*)nullptr, X4, E_EDGES, K3PAD, AGGH, 1);
    }
    // G4: (E,128)x(128,128) + c2 -> AGG (f32)
    {
      int waves = (E_EDGES / 16) * (H_DIM / 64);
      gemm_bf16_wmma<<<cdiv(waves, 8), BT, 0, stream>>>(
          X4, A2t, c2, AGG, (unsigned short*)nullptr, E_EDGES, AGGH, H_DIM, 0);
    }

    // out = mask*agg + (1-mask)*old
    blend_kernel<<<cdiv(E_EDGES * H_DIM, BT), BT, 0, stream>>>(AGG, oldst, mask_e, outHalf);
  };

  // v branch: aggregates over variables, updates function_state (second output half)
  run_branch(dec_var, var_idx,
             (const float*)d_in[12], (const float*)d_in[13],
             (const float*)d_in[14], (const float*)d_in[15],
             (const float*)d_in[16], (const float*)d_in[17],
             (const float*)d_in[18], (const float*)d_in[19],
             function_state, out + (size_t)E_EDGES * H_DIM);

  // f branch: aggregates over functions, updates variable_state (first output half)
  run_branch(dec_fun, fun_idx,
             (const float*)d_in[20], (const float*)d_in[21],
             (const float*)d_in[22], (const float*)d_in[23],
             (const float*)d_in[24], (const float*)d_in[25],
             (const float*)d_in[26], (const float*)d_in[27],
             variable_state, out);
}
